// Informer_58909771432891
// MI455X (gfx1250) — compile-verified, matrix operand order checked by reading, not run on hardware
//
#include <hip/hip_runtime.h>
#include <hip/hip_fp16.h>
#include <math.h>

typedef __attribute__((ext_vector_type(16))) _Float16 v16h;
typedef __attribute__((ext_vector_type(8)))  _Float16 v8h;
typedef __attribute__((ext_vector_type(4)))  _Float16 v4h;
typedef __attribute__((ext_vector_type(8)))  float    v8f;
typedef __attribute__((ext_vector_type(4)))  float    v4f;

// ---------------------------------------------------------------- helpers
__device__ __forceinline__ float wave_sum32(float v){
#pragma unroll
  for (int off = 16; off > 0; off >>= 1) v += __shfl_xor(v, off, 32);
  return v;
}
__device__ __forceinline__ float gelu_exact(float x){
  return 0.5f * x * (1.0f + erff(x * 0.70710678118654752f));
}
__device__ __forceinline__ unsigned mix3(unsigned a, unsigned b, unsigned c){
  unsigned x = a*0x9E3779B1u ^ (b+0x85EBCA77u)*0xC2B2AE3Du ^ (c*0x27D4EB2Fu + 0x165667B1u);
  x ^= x>>15; x *= 0x2C1B3C6Du; x ^= x>>12; x *= 0x297A2D39u; x ^= x>>15;
  return x;
}

enum { GF_BIAS=1, GF_RES=2, GF_GELU=4 };

// ------------------------------------------------- WMMA f16 GEMM (the workhorse)
// C[M,N](f32) = A x B, f32 storage, f16 WMMA compute, f32 accumulate.
// AMODE 0: A[M,K] row-major.
// AMODE 1: im2col for circular 3-tap conv: K==1536, A is [M,512] activations,
//          column k' = ci*3+r reads A[rowshift(m, r-1)*512 + ci], rows wrap
//          within sequences of 2^lsLog2.
// BMODE 0: B element (k,n) = Bm[n*ldbN + k]   (contiguous in k)
// BMODE 1: B element (k,n) = Bm[k*ldbK + n]   (contiguous in n)
// Requires M%64==0, N%128==0, K%32==0 (true for every call here).
template<int AMODE, int BMODE>
__global__ __launch_bounds__(256)
void gemm_f16wmma(const float* __restrict__ A, const float* __restrict__ Bm,
                  const float* __restrict__ bias, const float* __restrict__ Res,
                  float* __restrict__ C,
                  int M, int N, int K, int ldbK, int ldbN,
                  int flags, int lsLog2)
{
  __shared__ _Float16 As[64][32];    // [m][k]
  __shared__ _Float16 Bs[128][32];   // [n][k]  (transposed in LDS)
  const int tid  = threadIdx.x;
  const int lane = tid & 31, wv = tid >> 5;
  const int wm = wv >> 2, wn = wv & 3;            // 2 x 4 wave grid
  const int blockM = blockIdx.y * 64;
  const int blockN = blockIdx.x * 128;
  const int Ls = 1 << lsLog2;
  const int mask = Ls - 1;

  v8f acc[2][2] = {};

  for (int k0 = 0; k0 < K; k0 += 32) {
    // ---- stage A: 64x32 (2 half4 chunks per thread)
#pragma unroll
    for (int j = 0; j < 2; ++j) {
      int s = tid + j * 256;
      int r = s >> 3, c = (s & 7) * 4;
      int m = blockM + r;
      if (AMODE == 0) {
        v4f av = *(const v4f*)(A + (size_t)((unsigned)m * (unsigned)K + (unsigned)(k0 + c)));
        v4h ah = { (_Float16)av.x, (_Float16)av.y, (_Float16)av.z, (_Float16)av.w };
        *(v4h*)(&As[r][c]) = ah;
      } else {
        _Float16 hv[4];
        int t = m & mask, mbase = m & ~mask;
#pragma unroll
        for (int q = 0; q < 4; ++q) {
          int kq = k0 + c + q;
          int ci = kq / 3, rr = kq - 3 * ci;          // tap r in {0,1,2}
          int row = mbase | ((t + rr - 1 + Ls) & mask);
          hv[q] = (_Float16)A[(size_t)((unsigned)row * 512u + (unsigned)ci)];
        }
        v4h ah = { hv[0], hv[1], hv[2], hv[3] };
        *(v4h*)(&As[r][c]) = ah;
      }
    }
    // ---- stage B transposed: 128 n-rows x 32 k
    if (BMODE == 0) {
      // contiguous in k: vector load, vector store
#pragma unroll
      for (int j = 0; j < 4; ++j) {
        int s = tid + j * 256;
        int nn = s >> 3, c = (s & 7) * 4;
        v4f bv4 = *(const v4f*)(Bm + (size_t)((unsigned)(blockN + nn) * (unsigned)ldbN + (unsigned)(k0 + c)));
        v4h bh = { (_Float16)bv4.x, (_Float16)bv4.y, (_Float16)bv4.z, (_Float16)bv4.w };
        *(v4h*)(&Bs[nn][c]) = bh;
      }
    } else {
      // contiguous in n: 4x4 micro-tile register transpose
      int kq = tid >> 5;                       // 0..7  -> k = kq*4
      int nq = tid & 31;                       // 0..31 -> n = nq*4
      const float* bp = Bm + (size_t)((unsigned)(k0 + kq * 4) * (unsigned)ldbK + (unsigned)(blockN + nq * 4));
      v4f r0 = *(const v4f*)(bp);
      v4f r1 = *(const v4f*)(bp + ldbK);
      v4f r2 = *(const v4f*)(bp + 2 * ldbK);
      v4f r3 = *(const v4f*)(bp + 3 * ldbK);
      v4h c0 = { (_Float16)r0.x, (_Float16)r1.x, (_Float16)r2.x, (_Float16)r3.x };
      v4h c1 = { (_Float16)r0.y, (_Float16)r1.y, (_Float16)r2.y, (_Float16)r3.y };
      v4h c2 = { (_Float16)r0.z, (_Float16)r1.z, (_Float16)r2.z, (_Float16)r3.z };
      v4h c3 = { (_Float16)r0.w, (_Float16)r1.w, (_Float16)r2.w, (_Float16)r3.w };
      *(v4h*)(&Bs[nq * 4 + 0][kq * 4]) = c0;
      *(v4h*)(&Bs[nq * 4 + 1][kq * 4]) = c1;
      *(v4h*)(&Bs[nq * 4 + 2][kq * 4]) = c2;
      *(v4h*)(&Bs[nq * 4 + 3][kq * 4]) = c3;
    }
    __syncthreads();

    // ---- fragments: two 16B ds loads each, per ISA 16-bit A/B layout
    const int l15 = lane & 15;
    const int kb  = (lane >> 4) * 8;
    v16h af[2], bf[2];
#pragma unroll
    for (int i = 0; i < 2; ++i) {
      const _Float16* ar = &As[wm * 32 + i * 16 + l15][0];
      v8h lo = *(const v8h*)(ar + kb);
      v8h hi = *(const v8h*)(ar + 16 + kb);
      af[i] = __builtin_shufflevector(lo, hi, 0,1,2,3,4,5,6,7,8,9,10,11,12,13,14,15);
    }
#pragma unroll
    for (int i = 0; i < 2; ++i) {
      const _Float16* br = &Bs[wn * 32 + i * 16 + l15][0];
      v8h lo = *(const v8h*)(br + kb);
      v8h hi = *(const v8h*)(br + 16 + kb);
      bf[i] = __builtin_shufflevector(lo, hi, 0,1,2,3,4,5,6,7,8,9,10,11,12,13,14,15);
    }
#pragma unroll
    for (int i = 0; i < 2; ++i)
#pragma unroll
      for (int j = 0; j < 2; ++j)
        acc[i][j] = __builtin_amdgcn_wmma_f32_16x16x32_f16(
            false, af[i], false, bf[j], (short)0, acc[i][j], false, false);
    __syncthreads();
  }

#pragma unroll
  for (int i = 0; i < 2; ++i)
#pragma unroll
    for (int j = 0; j < 2; ++j) {
      int rowBase = blockM + wm * 32 + i * 16 + (lane >> 4) * 8;
      int col     = blockN + wn * 32 + j * 16 + (lane & 15);
      float bcol  = (flags & GF_BIAS) ? bias[col] : 0.0f;
#pragma unroll
      for (int r = 0; r < 8; ++r) {
        unsigned off = (unsigned)(rowBase + r) * (unsigned)N + (unsigned)col;
        float v = acc[i][j][r] + bcol;
        if (flags & GF_RES)  v += Res[off];
        if (flags & GF_GELU) v = gelu_exact(v);
        C[off] = v;
      }
    }
}

// ------------------------------------------------- embedding: h = x @ Wemb + b
__global__ void embed_kernel(const float* __restrict__ x, const float* __restrict__ W,
                             const float* __restrict__ b, float* __restrict__ out, int M)
{
  int gid = blockIdx.x * 256 + threadIdx.x;
  if (gid >= M * 512) return;
  int n = gid & 511, m = gid >> 9;
  const float* xr = x + (size_t)m * 7;
  float s = b[n];
#pragma unroll
  for (int k = 0; k < 7; ++k) s += xr[k] * W[k * 512 + n];
  out[(size_t)m * 512 + n] = s;
}

// ------------------------------------------------- layernorm over 512 cols, wave/row
__global__ __launch_bounds__(256)
void ln_kernel(const float* __restrict__ in, const float* __restrict__ g,
               const float* __restrict__ b, float* __restrict__ out, int M)
{
  int wv = threadIdx.x >> 5, lane = threadIdx.x & 31;
  int row = blockIdx.x * 8 + wv;
  if (row >= M) return;
  const float* p = in + (size_t)row * 512;
  float vals[16]; float s = 0.f;
#pragma unroll
  for (int i = 0; i < 16; ++i) { vals[i] = p[lane + i * 32]; s += vals[i]; }
  float mean = wave_sum32(s) * (1.0f / 512.0f);
  float vs = 0.f;
#pragma unroll
  for (int i = 0; i < 16; ++i) { float d = vals[i] - mean; vs += d * d; }
  float inv = rsqrtf(wave_sum32(vs) * (1.0f / 512.0f) + 1e-5f);
  float* o = out + (size_t)row * 512;
#pragma unroll
  for (int i = 0; i < 16; ++i) {
    int c = lane + i * 32;
    o[c] = (vals[i] - mean) * inv * g[c] + b[c];
  }
}

// ------------------------------------------------- prob-sparse sampling scores
__global__ void sample_score_kernel(const float* __restrict__ Q, const float* __restrict__ Kb,
                                    float* __restrict__ msc, int Lq, int sampleK, int layer)
{
  int gid = blockIdx.x * 256 + threadIdx.x;
  int total = 16 * 8 * Lq;
  if (gid >= total) return;
  int l = gid % Lq, rem = gid / Lq;
  int h = rem & 7, b = rem >> 3;
  const float* qp = Q + ((size_t)(b * Lq + l)) * 512 + h * 64;
  float qr[64];
#pragma unroll
  for (int d = 0; d < 64; ++d) qr[d] = qp[d];
  float mx = -3.4e38f, sm = 0.f;
  for (int s = 0; s < sampleK; ++s) {
    unsigned idx = mix3((unsigned)l, (unsigned)s, (unsigned)layer) & (unsigned)(Lq - 1);
    const float* kp = Kb + ((size_t)(b * Lq + (int)idx)) * 512 + h * 64;
    float dot = 0.f;
#pragma unroll
    for (int d = 0; d < 64; ++d) dot += qr[d] * kp[d];
    mx = fmaxf(mx, dot); sm += dot;
  }
  msc[((size_t)(b * 8 + h)) * Lq + l] = mx - sm / (float)Lq;
}

// ------------------------------------------------- iterative top-k per (b,h)
__global__ __launch_bounds__(256)
void topk_kernel(float* __restrict__ msc, int* __restrict__ topidx, int Lq, int nTop)
{
  __shared__ float sv[256]; __shared__ int si[256];
  int tid = threadIdx.x;
  float* Mrow = msc + (size_t)blockIdx.x * Lq;
  for (int u = 0; u < nTop; ++u) {
    float best = -3.4e38f; int bi = 0x7fffffff;
    for (int l = tid; l < Lq; l += 256) {
      float v = Mrow[l];
      if (v > best || (v == best && l < bi)) { best = v; bi = l; }
    }
    sv[tid] = best; si[tid] = bi; __syncthreads();
    for (int s = 128; s > 0; s >>= 1) {
      if (tid < s) {
        if (sv[tid+s] > sv[tid] || (sv[tid+s] == sv[tid] && si[tid+s] < si[tid])) {
          sv[tid] = sv[tid+s]; si[tid] = si[tid+s];
        }
      }
      __syncthreads();
    }
    if (tid == 0) { topidx[blockIdx.x * nTop + u] = si[0]; Mrow[si[0]] = -3.4e38f; }
    __syncthreads();
  }
}

// ------------------------------------------------- v mean over Lk per (b,h,d)
__global__ void vmean_kernel(const float* __restrict__ V, float* __restrict__ vm, int Lk)
{
  int gid = blockIdx.x * 256 + threadIdx.x;
  if (gid >= 16 * 512) return;            // gid = b*512 + h*64 + d
  int b = gid >> 9, c = gid & 511;
  float s = 0.f;
  for (int l = 0; l < Lk; ++l) s += V[((size_t)(b * Lk + l)) * 512 + c];
  vm[gid] = s / (float)Lk;
}

// ------------------------------------------------- ctx init (broadcast v-mean)
__global__ void ctx_fill_kernel(const float* __restrict__ vm, float* __restrict__ Cbuf,
                                int total, int lsLog2)
{
  int gid = blockIdx.x * 256 + threadIdx.x;
  if (gid >= total) return;
  int c = gid & 511, m = gid >> 9, b = m >> lsLog2;
  Cbuf[gid] = vm[(b << 9) | c];
}

// ------------------------------------------------- attention update for top queries
__global__ __launch_bounds__(256)
void attn_upd_kernel(const float* __restrict__ Q, const float* __restrict__ Kb,
                     const float* __restrict__ V, const int* __restrict__ topidx,
                     float* __restrict__ Cbuf, int Lk, int nTop)
{
  __shared__ float qs[64];
  __shared__ float red[256];
  __shared__ float pbuf[256];
  int tid = threadIdx.x;
  int b = blockIdx.x / (8 * nTop);
  int rem = blockIdx.x % (8 * nTop);
  int h = rem / nTop, u = rem % nTop;
  int l = topidx[(b * 8 + h) * nTop + u];
  if (tid < 64) qs[tid] = Q[((size_t)(b * Lk + l)) * 512 + h * 64 + tid];
  __syncthreads();

  float lmax = -3.4e38f;                     // pass 1: global max score
  for (int k = tid; k < Lk; k += 256) {
    const float* kp = Kb + ((size_t)(b * Lk + k)) * 512 + h * 64;
    float dot = 0.f;
#pragma unroll
    for (int d = 0; d < 64; ++d) dot += qs[d] * kp[d];
    lmax = fmaxf(lmax, dot * 0.125f);
  }
  red[tid] = lmax; __syncthreads();
  for (int s = 128; s > 0; s >>= 1) { if (tid < s) red[tid] = fmaxf(red[tid], red[tid+s]); __syncthreads(); }
  float gmax = red[0]; __syncthreads();

  float accd = 0.f, ssum = 0.f;              // pass 2: exp + weighted V
  for (int k0 = 0; k0 < Lk; k0 += 256) {
    int k = k0 + tid;
    float e = 0.f;
    if (k < Lk) {
      const float* kp = Kb + ((size_t)(b * Lk + k)) * 512 + h * 64;
      float dot = 0.f;
#pragma unroll
      for (int d = 0; d < 64; ++d) dot += qs[d] * kp[d];
      e = expf(dot * 0.125f - gmax);
    }
    pbuf[tid] = e; ssum += e;
    __syncthreads();
    if (tid < 64) {
      int lim = (Lk - k0 < 256) ? (Lk - k0) : 256;
      for (int j = 0; j < lim; ++j)
        accd += pbuf[j] * V[((size_t)(b * Lk + k0 + j)) * 512 + h * 64 + tid];
    }
    __syncthreads();
  }
  red[tid] = ssum; __syncthreads();
  for (int s = 128; s > 0; s >>= 1) { if (tid < s) red[tid] += red[tid+s]; __syncthreads(); }
  float tot = red[0];
  if (tid < 64) Cbuf[((size_t)(b * Lk + l)) * 512 + h * 64 + tid] = accd / tot;
}

// ------------------------------------------------- conv epilogue: bias+BN+ELU+maxpool/2
__global__ void conv_pool_kernel(const float* __restrict__ Y, const float* __restrict__ cb,
                                 const float* __restrict__ bng, const float* __restrict__ bnb,
                                 float* __restrict__ out, int Ls)
{
  int Lout = Ls >> 1;
  int gid = blockIdx.x * 256 + threadIdx.x;
  int total = 16 * Lout * 512;
  if (gid >= total) return;
  int c = gid & 511, rest = gid >> 9;
  int tp = rest % Lout, b = rest / Lout;
  float g = bng[c] * rsqrtf(1.0f + 1e-5f), bb = bnb[c], bias = cb[c];
  float best = -3.4e38f;
#pragma unroll
  for (int r = -1; r <= 1; ++r) {
    int t = 2 * tp + r;
    if (t >= 0 && t < Ls) {
      float yv = Y[((size_t)(b * Ls + t)) * 512 + c] + bias;
      yv = yv * g + bb;
      yv = (yv > 0.f) ? yv : (expf(yv) - 1.0f);
      best = fmaxf(best, yv);
    }
  }
  out[((size_t)(b * Lout + tp)) * 512 + c] = best;
}

// ------------------------------------------------- final projection (N=1), wave/row
__global__ __launch_bounds__(256)
void final_proj_kernel(const float* __restrict__ Hn, const float* __restrict__ pw,
                       const float* __restrict__ pb, float* __restrict__ out)
{
  int wid = (blockIdx.x * 256 + threadIdx.x) >> 5;
  int lane = threadIdx.x & 31;
  if (wid >= 16 * 47) return;
  int b = wid / 47, t = wid % 47;
  const float* hr = Hn + ((size_t)(b * 1024 + t)) * 512;
  float s = 0.f;
#pragma unroll
  for (int i = 0; i < 16; ++i) s += hr[lane + i * 32] * pw[lane + i * 32];
  s = wave_sum32(s);
  if (lane == 0) out[wid] = s + pb[0];
}

// ------------------------------------------------- host orchestration
static inline void launch_gemm(int amode, int bmode,
                               const float* A, const float* B, const float* bias,
                               const float* Res, float* C, int M, int N, int K,
                               int ldbK, int ldbN, int flags, int lsLog2,
                               hipStream_t s)
{
  dim3 g((unsigned)(N / 128), (unsigned)(M / 64));
  if (amode == 1)
    gemm_f16wmma<1,0><<<g, 256, 0, s>>>(A, B, bias, Res, C, M, N, K, ldbK, ldbN, flags, lsLog2);
  else if (bmode == 0)
    gemm_f16wmma<0,0><<<g, 256, 0, s>>>(A, B, bias, Res, C, M, N, K, ldbK, ldbN, flags, lsLog2);
  else
    gemm_f16wmma<0,1><<<g, 256, 0, s>>>(A, B, bias, Res, C, M, N, K, ldbK, ldbN, flags, lsLog2);
}

extern "C" void kernel_launch(void* const* d_in, const int* in_sizes, int n_in,
                              void* d_out, int out_size, void* d_ws, size_t ws_size,
                              hipStream_t stream) {
  (void)in_sizes; (void)n_in; (void)out_size; (void)ws_size;
  const float* x     = (const float*)d_in[0];
  const float* embW  = (const float*)d_in[1];
  const float* embB  = (const float*)d_in[2];
  const float* Wq    = (const float*)d_in[3];
  const float* bq    = (const float*)d_in[4];
  const float* Wk    = (const float*)d_in[5];
  const float* bk    = (const float*)d_in[6];
  const float* Wv    = (const float*)d_in[7];
  const float* bv    = (const float*)d_in[8];
  const float* Wo    = (const float*)d_in[9];
  const float* bo    = (const float*)d_in[10];
  const float* W1    = (const float*)d_in[11];
  const float* b1    = (const float*)d_in[12];
  const float* W2    = (const float*)d_in[13];
  const float* b2    = (const float*)d_in[14];
  const float* ln1g  = (const float*)d_in[15];
  const float* ln1b  = (const float*)d_in[16];
  const float* ln2g  = (const float*)d_in[17];
  const float* ln2b  = (const float*)d_in[18];
  const float* convW = (const float*)d_in[19];
  const float* convB = (const float*)d_in[20];
  const float* bng   = (const float*)d_in[21];
  const float* bnb   = (const float*)d_in[22];
  const float* flng  = (const float*)d_in[23];
  const float* flnb  = (const float*)d_in[24];
  const float* projW = (const float*)d_in[25];
  const float* projB = (const float*)d_in[26];
  float* out = (float*)d_out;

  char* base = (char*)d_ws;
  const size_t BUF = (size_t)16 * 4096 * 512 * sizeof(float);   // 128 MB
  float* pA = (float*)(base);
  float* pQ = (float*)(base + 1 * BUF);
  float* pK = (float*)(base + 2 * BUF);
  float* pV = (float*)(base + 3 * BUF);
  float* pC = (float*)(base + 4 * BUF);
  float* msc = (float*)(base + 5 * BUF);                         // 2 MB
  int*   tix = (int*)  (base + 5 * BUF + (size_t)16 * 8 * 4096 * 4);
  float* vm  = (float*)(base + 5 * BUF + (size_t)16 * 8 * 4096 * 4 + 65536);

  // embedding
  {
    int M = 16 * 4096;
    embed_kernel<<<(M * 512 + 255) / 256, 256, 0, stream>>>(x, embW, embB, pA, M);
  }

  int Ls = 4096;
  for (int layer = 0; layer < 3; ++layer) {
    const int M = 16 * Ls;
    const int lsLog2 = (Ls == 4096) ? 12 : (Ls == 2048) ? 11 : 10;
    const int sk = (Ls == 4096) ? 45 : (Ls == 2048) ? 40 : 35;   // 5*ceil(ln(L))

    // Q, K, V projections (BMODE 1: weights stored [k][n])
    launch_gemm(0,1, pA, Wq + (size_t)layer*512*512, bq + layer*512, nullptr, pQ,
                M, 512, 512, 512, 1, GF_BIAS, lsLog2, stream);
    launch_gemm(0,1, pA, Wk + (size_t)layer*512*512, bk + layer*512, nullptr, pK,
                M, 512, 512, 512, 1, GF_BIAS, lsLog2, stream);
    launch_gemm(0,1, pA, Wv + (size_t)layer*512*512, bv + layer*512, nullptr, pV,
                M, 512, 512, 512, 1, GF_BIAS, lsLog2, stream);

    // prob-sparse attention
    sample_score_kernel<<<(16*8*Ls + 255)/256, 256, 0, stream>>>(pQ, pK, msc, Ls, sk, layer);
    topk_kernel<<<16*8, 256, 0, stream>>>(msc, tix, Ls, sk);
    vmean_kernel<<<(16*512 + 255)/256, 256, 0, stream>>>(pV, vm, Ls);
    ctx_fill_kernel<<<(M*512 + 255)/256, 256, 0, stream>>>(vm, pC, M*512, lsLog2);
    attn_upd_kernel<<<16*8*sk, 256, 0, stream>>>(pQ, pK, pV, tix, pC, Ls, sk);

    // output projection + residual, then LN1
    launch_gemm(0,1, pC, Wo + (size_t)layer*512*512, bo + layer*512, pA, pQ,
                M, 512, 512, 512, 1, GF_BIAS | GF_RES, lsLog2, stream);
    ln_kernel<<<(M + 7)/8, 256, 0, stream>>>(pQ, ln1g + layer*512, ln1b + layer*512, pA, M);

    // FFN (chunked: intermediate 2048-wide lives in pK; BMODE 0: weights [n][k])
    {
      const int CH = 8192;
      float* tmp = pK;
      for (int c0 = 0; c0 < M; c0 += CH) {
        int cm = (M - c0 < CH) ? (M - c0) : CH;
        launch_gemm(0,0, pA + (size_t)c0*512, W1 + (size_t)layer*2048*512, b1 + layer*2048,
                    nullptr, tmp, cm, 2048, 512, 1, 512, GF_BIAS | GF_GELU, lsLog2, stream);
        launch_gemm(0,0, tmp, W2 + (size_t)layer*512*2048, b2 + layer*512,
                    pA + (size_t)c0*512, pC + (size_t)c0*512,
                    cm, 512, 2048, 1, 2048, GF_BIAS | GF_RES, lsLog2, stream);
      }
    }
    ln_kernel<<<(M + 7)/8, 256, 0, stream>>>(pC, ln2g + layer*512, ln2b + layer*512, pA, M);

    // conv (fused im2col, K=1536, conv_W is [co][ci*3+r] -> transposed-B fast path)
    if (layer < 2) {
      launch_gemm(1,0, pA, convW + (size_t)layer*512*512*3, nullptr, nullptr, pQ,
                  M, 512, 1536, 1, 1536, 0, lsLog2, stream);
      conv_pool_kernel<<<(16*(Ls/2)*512 + 255)/256, 256, 0, stream>>>(
          pQ, convB + layer*512, bng + layer*512, bnb + layer*512, pC, Ls);
      float* t2 = pA; pA = pC; pC = t2;
      Ls >>= 1;
    }
  }

  // final LN + projection, slice [:, :47, :]
  ln_kernel<<<(16*1024 + 7)/8, 256, 0, stream>>>(pA, flng, flnb, pC, 16*1024);
  final_proj_kernel<<<(16*47*32 + 255)/256, 256, 0, stream>>>(pC, projW, projB, out);
}
